// GVPPocketClassifier_16887811407938
// MI455X (gfx1250) — matrix operand order repaired and checked by reading, not compiled, not run on hardware
//
#include <hip/hip_runtime.h>
#include <hip/hip_bf16.h>

typedef __attribute__((ext_vector_type(16))) _Float16 v16h;
typedef __attribute__((ext_vector_type(8)))  float    v8f;

#define NNODES 32768
#define NEDGES 524288
#define NBATCH 64
#define NLAYER 4
#define ESMD   1280
#define SDIM   128
#define VDIM   16
#define EHDIM  64
#define GAMMA_C (1.0f/(0.5625f+1e-8f))   // 1/((12/16)^2 + 1e-8)
#define RBF_STEP 0.8f                    // linspace(0,12,16) spacing
#define MSG_K 384                        // 354 padded to 32
#define OUT_K 224                        // 203 padded to 32
#define UPD_K 288

__device__ __forceinline__ float silu_f(float x){ return x / (1.f + __expf(-x)); }
__device__ __forceinline__ float sigm_f(float x){ return 1.f / (1.f + __expf(-x)); }

// A-fragment swizzle for V_WMMA_F32_16X16X32_F16 (16-bit A, 16x32 per k-tile):
// lane<16 holds row M=lane, halves[0..7]=K0..7, halves[8..15]=K16..23
// lane>=16 holds row M=lane-16, halves[0..7]=K8..15, halves[8..15]=K24..31
// NOTE: any 8-aligned K-run of 8 elements is CONTIGUOUS in this layout.
__device__ __forceinline__ int a_swz(int row, int k){
  int kk = k & 31;
  int lane = row + (((kk >> 3) & 1) << 4);
  int idx  = (kk & 7) + ((kk >> 4) << 3);
  return ((k >> 5) * 32 + lane) * 16 + idx;
}

// 8 consecutive f32 -> 8 f16 -> one 16B LDS store (dst must be 16B aligned)
__device__ __forceinline__ void store8_f32tof16(_Float16* dst, const float* src){
  float4 a = *(const float4*)(src);
  float4 b = *(const float4*)(src + 4);
  __align__(16) _Float16 h[8] = {
    (_Float16)a.x, (_Float16)a.y, (_Float16)a.z, (_Float16)a.w,
    (_Float16)b.x, (_Float16)b.y, (_Float16)b.z, (_Float16)b.w };
  *(uint4*)dst = *(const uint4*)h;
}

// ---------------- weight packing: f32 row-major (K,N) -> f16 WMMA B fragments
// B layout per 32x16 k-tile: lane l -> column l%16; lane<16: halves[0..7]=K0..7,
// halves[8..15]=K16..23; lane>=16: halves[0..7]=K8..15, halves[8..15]=K24..31.
__global__ void pack_b(const float* __restrict__ W, _Float16* __restrict__ out,
                       int K, int N, int Kpad){
  int total = Kpad * N;
  int Ntiles = N >> 4;
  for (int i = blockIdx.x*blockDim.x + threadIdx.x; i < total; i += gridDim.x*blockDim.x){
    int h = i & 15; int lane = (i >> 4) & 31; int blk = i >> 9;
    int nt = blk % Ntiles; int kt = blk / Ntiles;
    int k = kt*32 + ((lane >> 4) << 3) + (h & 7) + ((h >> 3) << 4);
    int n = nt*16 + (lane & 15);
    float val = (k < K) ? W[(long)k*N + n] : 0.f;
    out[i] = (_Float16)val;
  }
}

__global__ void zero_f32(float* p, long n){
  long i = (long)blockIdx.x*blockDim.x + threadIdx.x;
  if (i < n) p[i] = 0.f;
}

// ---------------- node encoder: big GEMM  tmp = x_esm @ ne_esm_w  (N x 1280 x 128)
__global__ __launch_bounds__(256) void gemm_esm(const float* __restrict__ X,
                                                const _Float16* __restrict__ Bp,
                                                float* __restrict__ out){
  __shared__ __align__(32) _Float16 aS[16*256];
  const int tid = threadIdx.x, lane = tid & 31, wave = tid >> 5;
  const long row0 = (long)blockIdx.x * 16;
  v8f acc = {};
  for (int k0 = 0; k0 < ESMD; k0 += 256){
    // 16 rows x 256 K in 8-element chunks: 512 chunks, 2 per thread
    for (int c = tid; c < 512; c += 256){
      int r = c >> 5, g = c & 31;
      store8_f32tof16(aS + a_swz(r, g*8), X + (row0 + r)*ESMD + k0 + g*8);
    }
    __syncthreads();
    #pragma unroll
    for (int kt = 0; kt < 8; ++kt){
      v16h a = *(const v16h*)(aS + (kt*32 + lane)*16);
      long ktg = (k0 >> 5) + kt;
      v16h b = *(const v16h*)(Bp + ((ktg*8 + wave)*32 + lane)*16);
      acc = __builtin_amdgcn_wmma_f32_16x16x32_f16(false, a, false, b, (short)0, acc, false, false);
    }
    __syncthreads();
  }
  int col = wave*16 + (lane & 15);
  int rb  = (lane >> 4) * 8;
  #pragma unroll
  for (int r = 0; r < 8; ++r)
    out[(row0 + rb + r)*SDIM + col] = acc[r];
}

// ---------------- bias+LN+SiLU on esm, build xcat (N x 224 f16, zero padded)
__global__ __launch_bounds__(256) void node_feat(const float* __restrict__ tmp,
  const float* __restrict__ bias, const float* __restrict__ g, const float* __restrict__ bln,
  const float* __restrict__ reschem, const float* __restrict__ role,
  const float* __restrict__ misc, const float* __restrict__ dist,
  _Float16* __restrict__ xcat){
  long node = (long)blockIdx.x*8 + (threadIdx.x >> 5);
  int lane = threadIdx.x & 31;
  float vals[4]; float sum = 0.f;
  #pragma unroll
  for (int i = 0; i < 4; ++i){
    vals[i] = tmp[node*128 + lane*4 + i] + bias[lane*4 + i];
    sum += vals[i];
  }
  for (int o = 16; o > 0; o >>= 1) sum += __shfl_xor(sum, o, 32);
  float mu = sum / 128.f;
  float var = 0.f;
  #pragma unroll
  for (int i = 0; i < 4; ++i){ float d = vals[i]-mu; var += d*d; }
  for (int o = 16; o > 0; o >>= 1) var += __shfl_xor(var, o, 32);
  float rstd = rsqrtf(var/128.f + 1e-5f);
  _Float16* xr = xcat + node*OUT_K;
  #pragma unroll
  for (int i = 0; i < 4; ++i){
    int c = lane*4 + i;
    xr[c] = (_Float16)silu_f((vals[i]-mu)*rstd*g[c] + bln[c]);
  }
  if (lane < 6) xr[128+lane] = (_Float16)reschem[node*6+lane];
  if (lane < 2) xr[134+lane] = (_Float16)role[node*2+lane];
  if (lane < 3) xr[136+lane] = (_Float16)misc[node*3+lane];
  for (int j = lane; j < 64; j += 32){
    int di = j >> 4, ci = j & 15;
    float t = dist[node*4+di] - RBF_STEP*ci;
    xr[139+j] = (_Float16)__expf(-GAMMA_C*t*t);
  }
  for (int j = lane; j < OUT_K-203; j += 32) xr[203+j] = (_Float16)0.f;
}

// ---------------- s = silu(ln(xcat @ ne_out_w + b))
__global__ __launch_bounds__(256) void gemm_node_out(const _Float16* __restrict__ xcat,
  const _Float16* __restrict__ Bp, const float* __restrict__ bias,
  const float* __restrict__ g, const float* __restrict__ bln, float* __restrict__ s){
  __shared__ __align__(32) _Float16 aS[16*OUT_K];
  __shared__ float dS[16*128];
  __shared__ float muS[16], rsS[16];
  const int tid = threadIdx.x, lane = tid & 31, wave = tid >> 5;
  const long row0 = (long)blockIdx.x * 16;
  // 16 rows x 28 chunks of 8 halves (f16 -> f16 straight copy)
  for (int c = tid; c < 512; c += 256){
    int r = c >> 5, gch = c & 31;
    if (gch < OUT_K/8)
      *(uint4*)(aS + a_swz(r, gch*8)) = *(const uint4*)(xcat + (row0 + r)*OUT_K + gch*8);
  }
  __syncthreads();
  v8f acc = {};
  #pragma unroll
  for (int kt = 0; kt < OUT_K/32; ++kt){
    v16h a = *(const v16h*)(aS + (kt*32 + lane)*16);
    v16h b = *(const v16h*)(Bp + (((long)kt*8 + wave)*32 + lane)*16);
    acc = __builtin_amdgcn_wmma_f32_16x16x32_f16(false, a, false, b, (short)0, acc, false, false);
  }
  int col = wave*16 + (lane & 15);
  int rb  = (lane >> 4) * 8;
  #pragma unroll
  for (int r = 0; r < 8; ++r) dS[(rb+r)*128 + col] = acc[r] + bias[col];
  __syncthreads();
  if (tid < 16){
    float* rowp = dS + tid*128;
    float mu = 0.f;  for (int c = 0; c < 128; ++c) mu += rowp[c];  mu /= 128.f;
    float var = 0.f; for (int c = 0; c < 128; ++c){ float d = rowp[c]-mu; var += d*d; }
    muS[tid] = mu; rsS[tid] = rsqrtf(var/128.f + 1e-5f);
  }
  __syncthreads();
  {
    int r = tid >> 4, c0 = (tid & 15) * 8;
    float mu = muS[r], rstd = rsS[r];
    float* so = s + (row0 + r)*128;
    #pragma unroll
    for (int i = 0; i < 8; ++i){
      int c = c0 + i;
      so[c] = silu_f((dS[r*128 + c] - mu)*rstd*g[c] + bln[c]);
    }
  }
}

__global__ void init_v(const float* __restrict__ xvec, const float* __restrict__ w,
                       float* __restrict__ v){
  long i = (long)blockIdx.x*blockDim.x + threadIdx.x;
  if (i >= (long)NNODES*48) return;
  long n = i / 48; int rest = i % 48; int vi = rest / 3, c = rest % 3;
  v[i] = xvec[n*3 + c] * w[vi];
}

// ---------------- edge encoder (VALU; tiny): edge_s (f16), edge_v, edge_len
__global__ __launch_bounds__(256) void edge_feat(const float* __restrict__ edist,
  const int* __restrict__ eidx, const float* __restrict__ pos,
  const float* __restrict__ eew, const float* __restrict__ eeb,
  const float* __restrict__ elng, const float* __restrict__ elnb,
  _Float16* __restrict__ edge_s, float* __restrict__ edge_v, float* __restrict__ edge_len){
  long e = (long)blockIdx.x*8 + (threadIdx.x >> 5);
  int lane = threadIdx.x & 31;
  int src = eidx[e], dst = eidx[NEDGES + e];
  int di = lane >> 4, ci = lane & 15;
  float t = edist[e*2 + di] - RBF_STEP*ci;
  float rbf = __expf(-GAMMA_C*t*t);
  float o0 = eeb[lane], o1 = eeb[32+lane];
  for (int k = 0; k < 32; ++k){
    float rk = __shfl(rbf, k, 32);
    o0 += rk * eew[k*64 + lane];
    o1 += rk * eew[k*64 + 32 + lane];
  }
  float sum = o0 + o1;
  for (int o = 16; o > 0; o >>= 1) sum += __shfl_xor(sum, o, 32);
  float mu = sum / 64.f;
  float var = (o0-mu)*(o0-mu) + (o1-mu)*(o1-mu);
  for (int o = 16; o > 0; o >>= 1) var += __shfl_xor(var, o, 32);
  float rstd = rsqrtf(var/64.f + 1e-5f);
  edge_s[e*64 + lane]      = (_Float16)silu_f((o0-mu)*rstd*elng[lane] + elnb[lane]);
  edge_s[e*64 + 32 + lane] = (_Float16)silu_f((o1-mu)*rstd*elng[32+lane] + elnb[32+lane]);
  if (lane == 0){
    float rx = pos[(long)dst*3+0]-pos[(long)src*3+0];
    float ry = pos[(long)dst*3+1]-pos[(long)src*3+1];
    float rz = pos[(long)dst*3+2]-pos[(long)src*3+2];
    float inv = rsqrtf(fmaxf(rx*rx+ry*ry+rz*rz, 1e-8f));
    float vx = rx*inv, vy = ry*inv, vz = rz*inv;
    edge_v[e*3+0] = vx; edge_v[e*3+1] = vy; edge_v[e*3+2] = vz;
    edge_len[e] = sqrtf(fmaxf(vx*vx+vy*vy+vz*vz, 1e-8f));
  }
}

// ---------------- per-layer message GVP + scatter (16 edges / block, 8 waves)
__global__ __launch_bounds__(256) void msg_kernel(
  const float* __restrict__ s, const float* __restrict__ v,
  const _Float16* __restrict__ edge_s, const float* __restrict__ edge_v,
  const float* __restrict__ edge_len, const int* __restrict__ eidx,
  const _Float16* __restrict__ B1, const float* __restrict__ b1,
  const _Float16* __restrict__ B2, const float* __restrict__ b2,
  const float* __restrict__ vw, const float* __restrict__ gw, const float* __restrict__ gb,
  float* __restrict__ agg_s, float* __restrict__ agg_v){
  __shared__ __align__(32) _Float16 aS[16*MSG_K];
  __shared__ __align__(32) _Float16 hS[16*128];
  __shared__ float soS[16*128];
  __shared__ float mvS[16][33][3];
  __shared__ float gateS[16][16];
  const int tid = threadIdx.x, lane = tid & 31, wave = tid >> 5;
  const long e0 = (long)blockIdx.x * 16;
  // stage m_v_in (v[src](16) | v[dst](16) | edge_v(1)), 3 floats each
  for (int i = tid; i < 16*33; i += 256){
    int e = i / 33, ch = i % 33;
    long eg = e0 + e;
    const float* p;
    if (ch < 16)      p = v + ((long)eidx[eg]*16 + ch)*3;
    else if (ch < 32) p = v + ((long)eidx[NEDGES+eg]*16 + (ch-16))*3;
    else              p = edge_v + eg*3;
    mvS[e][ch][0] = p[0]; mvS[e][ch][1] = p[1]; mvS[e][ch][2] = p[2];
  }
  // chunked gather of scalar A portion (k < 320): 16 threads per edge
  {
    int e = tid >> 4, gch = tid & 15;
    long eg = e0 + e;
    long srcn = eidx[eg], dstn = eidx[NEDGES + eg];
    store8_f32tof16(aS + a_swz(e, gch*8),        s + srcn*128 + gch*8);
    store8_f32tof16(aS + a_swz(e, 128 + gch*8),  s + dstn*128 + gch*8);
    if (gch < 8)
      *(uint4*)(aS + a_swz(e, 256 + gch*8)) = *(const uint4*)(edge_s + eg*64 + gch*8);
  }
  __syncthreads();
  // tail cols 320..383: edge_len | 33 vnorms | zero pad
  for (int i = tid; i < 16*64; i += 256){
    int e = i >> 6, kk = i & 63;
    int k = 320 + kk;
    float val;
    if (kk == 0) val = edge_len[e0 + e];
    else if (k < 354){
      int ch = k - 321;
      float x = mvS[e][ch][0], y = mvS[e][ch][1], z = mvS[e][ch][2];
      val = sqrtf(fmaxf(x*x+y*y+z*z, 1e-8f));
    } else val = 0.f;
    aS[a_swz(e,k)] = (_Float16)val;
  }
  __syncthreads();
  // GEMM1: (16 x 384) @ (384 x 128)
  v8f acc = {};
  #pragma unroll
  for (int kt = 0; kt < MSG_K/32; ++kt){
    v16h a = *(const v16h*)(aS + (kt*32 + lane)*16);
    v16h b = *(const v16h*)(B1 + (((long)kt*8 + wave)*32 + lane)*16);
    acc = __builtin_amdgcn_wmma_f32_16x16x32_f16(false, a, false, b, (short)0, acc, false, false);
  }
  {
    int col = wave*16 + (lane & 15);
    int rb  = (lane >> 4) * 8;
    float bc = b1[col];
    #pragma unroll
    for (int r = 0; r < 8; ++r)
      hS[a_swz(rb+r, col)] = (_Float16)silu_f(acc[r] + bc);
  }
  __syncthreads();
  // GEMM2: (16 x 128) @ (128 x 128)
  v8f acc2 = {};
  #pragma unroll
  for (int kt = 0; kt < 4; ++kt){
    v16h a = *(const v16h*)(hS + (kt*32 + lane)*16);
    v16h b = *(const v16h*)(B2 + (((long)kt*8 + wave)*32 + lane)*16);
    acc2 = __builtin_amdgcn_wmma_f32_16x16x32_f16(false, a, false, b, (short)0, acc2, false, false);
  }
  {
    int col = wave*16 + (lane & 15);
    int rb  = (lane >> 4) * 8;
    float bc = b2[col];
    #pragma unroll
    for (int r = 0; r < 8; ++r) soS[(rb+r)*128 + col] = acc2[r] + bc;
  }
  __syncthreads();
  {
    int e = tid >> 4, vi = tid & 15;
    float gsum = gb[vi];
    for (int k = 0; k < 128; ++k) gsum += soS[e*128 + k] * gw[k*16 + vi];
    gateS[e][vi] = sigm_f(gsum);
  }
  __syncthreads();
  // scatter m_s
  {
    int e = tid >> 4, c0 = (tid & 15) * 8;
    int dst = eidx[NEDGES + e0 + e];
    #pragma unroll
    for (int i = 0; i < 8; ++i)
      atomicAdd(&agg_s[(long)dst*128 + c0 + i], soS[e*128 + c0 + i]);
  }
  // scatter m_v
  {
    int e = tid >> 4, vi = tid & 15;
    int dst = eidx[NEDGES + e0 + e];
    float gate = gateS[e][vi];
    #pragma unroll
    for (int c = 0; c < 3; ++c){
      float p = 0.f;
      for (int ch = 0; ch < 33; ++ch) p += mvS[e][ch][c] * vw[ch*16 + vi];
      atomicAdd(&agg_v[((long)dst*16 + vi)*3 + c], p * gate);
    }
  }
}

// ---------------- per-layer node update GVP + residual LN (16 nodes / block)
__global__ __launch_bounds__(256) void upd_kernel(
  float* __restrict__ s, float* __restrict__ v,
  const float* __restrict__ agg_s, const float* __restrict__ agg_v,
  const _Float16* __restrict__ B1, const float* __restrict__ b1,
  const _Float16* __restrict__ B2, const float* __restrict__ b2,
  const float* __restrict__ vw, const float* __restrict__ gw, const float* __restrict__ gb,
  const float* __restrict__ lng, const float* __restrict__ lnb){
  __shared__ __align__(32) _Float16 aS[16*UPD_K];
  __shared__ __align__(32) _Float16 hS[16*128];
  __shared__ float soS[16*128];
  __shared__ float uvS[16][32][3];
  __shared__ float gateS[16][16];
  __shared__ float muS[16], rsS[16];
  const int tid = threadIdx.x, lane = tid & 31, wave = tid >> 5;
  const long n0 = (long)blockIdx.x * 16;
  for (int i = tid; i < 16*32; i += 256){
    int r = i >> 5, ch = i & 31;
    long n = n0 + r;
    const float* p = (ch < 16) ? (v + (n*16 + ch)*3) : (agg_v + (n*16 + (ch-16))*3);
    uvS[r][ch][0] = p[0]; uvS[r][ch][1] = p[1]; uvS[r][ch][2] = p[2];
  }
  // chunked scalar A portion (k < 256)
  {
    int r = tid >> 4, gch = tid & 15;
    long n = n0 + r;
    store8_f32tof16(aS + a_swz(r, gch*8),       s + n*128 + gch*8);
    store8_f32tof16(aS + a_swz(r, 128 + gch*8), agg_s + n*128 + gch*8);
  }
  __syncthreads();
  // vnorm cols 256..287
  for (int i = tid; i < 16*32; i += 256){
    int r = i >> 5, ch = i & 31;
    float x = uvS[r][ch][0], y = uvS[r][ch][1], z = uvS[r][ch][2];
    aS[a_swz(r, 256 + ch)] = (_Float16)sqrtf(fmaxf(x*x+y*y+z*z, 1e-8f));
  }
  __syncthreads();
  v8f acc = {};
  #pragma unroll
  for (int kt = 0; kt < UPD_K/32; ++kt){
    v16h a = *(const v16h*)(aS + (kt*32 + lane)*16);
    v16h b = *(const v16h*)(B1 + (((long)kt*8 + wave)*32 + lane)*16);
    acc = __builtin_amdgcn_wmma_f32_16x16x32_f16(false, a, false, b, (short)0, acc, false, false);
  }
  {
    int col = wave*16 + (lane & 15);
    int rb  = (lane >> 4) * 8;
    float bc = b1[col];
    #pragma unroll
    for (int r = 0; r < 8; ++r)
      hS[a_swz(rb+r, col)] = (_Float16)silu_f(acc[r] + bc);
  }
  __syncthreads();
  v8f acc2 = {};
  #pragma unroll
  for (int kt = 0; kt < 4; ++kt){
    v16h a = *(const v16h*)(hS + (kt*32 + lane)*16);
    v16h b = *(const v16h*)(B2 + (((long)kt*8 + wave)*32 + lane)*16);
    acc2 = __builtin_amdgcn_wmma_f32_16x16x32_f16(false, a, false, b, (short)0, acc2, false, false);
  }
  {
    int col = wave*16 + (lane & 15);
    int rb  = (lane >> 4) * 8;
    float bc = b2[col];
    #pragma unroll
    for (int r = 0; r < 8; ++r) soS[(rb+r)*128 + col] = acc2[r] + bc;
  }
  __syncthreads();
  {
    int r = tid >> 4, vi = tid & 15;
    float gsum = gb[vi];
    for (int k = 0; k < 128; ++k) gsum += soS[r*128 + k] * gw[k*16 + vi];
    gateS[r][vi] = sigm_f(gsum);
  }
  __syncthreads();
  // v += gate * (u_v_in @ vw), and add residual s into soS
  {
    int r = tid >> 4, vi = tid & 15;
    long n = n0 + r;
    float gate = gateS[r][vi];
    #pragma unroll
    for (int c = 0; c < 3; ++c){
      float p = 0.f;
      for (int ch = 0; ch < 32; ++ch) p += uvS[r][ch][c] * vw[ch*16 + vi];
      v[(n*16 + vi)*3 + c] += p * gate;
    }
    int c0 = vi * 8;
    #pragma unroll
    for (int i = 0; i < 8; ++i) soS[r*128 + c0 + i] += s[n*128 + c0 + i];
  }
  __syncthreads();
  if (tid < 16){
    float* rowp = soS + tid*128;
    float mu = 0.f;  for (int c = 0; c < 128; ++c) mu += rowp[c];  mu /= 128.f;
    float var = 0.f; for (int c = 0; c < 128; ++c){ float d = rowp[c]-mu; var += d*d; }
    muS[tid] = mu; rsS[tid] = rsqrtf(var/128.f + 1e-5f);
  }
  __syncthreads();
  {
    int r = tid >> 4, c0 = (tid & 15) * 8;
    long n = n0 + r;
    float mu = muS[r], rstd = rsS[r];
    #pragma unroll
    for (int i = 0; i < 8; ++i){
      int c = c0 + i;
      s[n*128 + c] = (soS[r*128 + c] - mu)*rstd*lng[c] + lnb[c];
    }
  }
}

// ---------------- pooling (segmented: batch is sorted, so reduce runs locally)
__device__ __forceinline__ unsigned fenc(float x){
  unsigned u = __float_as_uint(x);
  return (u & 0x80000000u) ? ~u : (u | 0x80000000u);
}
__device__ __forceinline__ float fdec(unsigned u){
  return (u & 0x80000000u) ? __uint_as_float(u & 0x7FFFFFFFu) : __uint_as_float(~u);
}

__global__ __launch_bounds__(256) void pool_kernel(const float* __restrict__ s,
    const int* __restrict__ batch, float* __restrict__ psum,
    float* __restrict__ pcnt, unsigned* __restrict__ pmax){
  // block covers 32 consecutive nodes; thread: col = t&127, 16-node run = t>>7
  long base = (long)blockIdx.x*32 + (threadIdx.x >> 7)*16;
  int c = threadIdx.x & 127;
  int cur_b = batch[base];
  float acc = 0.f; unsigned mx = 0u; int cnt = 0;
  for (int i = 0; i < 16; ++i){
    long n = base + i;
    int b = batch[n];
    if (b != cur_b){
      atomicAdd(&psum[cur_b*128 + c], acc);
      atomicMax(&pmax[cur_b*128 + c], mx);
      if (c == 0) atomicAdd(&pcnt[cur_b], (float)cnt);
      cur_b = b; acc = 0.f; mx = 0u; cnt = 0;
    }
    float val = s[n*128 + c];
    acc += val;
    unsigned e = fenc(val);
    mx = (mx > e) ? mx : e;
    ++cnt;
  }
  atomicAdd(&psum[cur_b*128 + c], acc);
  atomicMax(&pmax[cur_b*128 + c], mx);
  if (c == 0) atomicAdd(&pcnt[cur_b], (float)cnt);
}

__global__ __launch_bounds__(128) void head_kernel(const float* __restrict__ psum,
  const float* __restrict__ pcnt, const unsigned* __restrict__ pmax,
  const float* __restrict__ hw1, const float* __restrict__ hb1,
  const float* __restrict__ hw2, const float* __restrict__ hb2,
  const float* __restrict__ ew1, const float* __restrict__ eb1,
  const float* __restrict__ ew2, const float* __restrict__ eb2,
  float* __restrict__ out){
  __shared__ float emb[256];
  __shared__ float h1[128];
  int b = blockIdx.x, t = threadIdx.x;
  float cnt = fmaxf(pcnt[b], 1.f);
  emb[t] = psum[b*128 + t] / cnt;
  float m = fdec(pmax[b*128 + t]);
  if (!(fabsf(m) <= 3.0e38f)) m = 0.f;   // non-finite / NaN -> 0
  emb[128 + t] = m;
  __syncthreads();
  float acc = hb1[t];
  for (int k = 0; k < 256; ++k) acc += emb[k] * hw1[k*128 + t];
  h1[t] = silu_f(acc);
  __syncthreads();
  if (t < 8){
    float o = hb2[t];
    for (int k = 0; k < 128; ++k) o += h1[k] * hw2[k*8 + t];
    out[b*8 + t] = o;
  }
  __syncthreads();
  float acc2 = eb1[t];
  for (int k = 0; k < 256; ++k) acc2 += emb[k] * ew1[k*128 + t];
  h1[t] = silu_f(acc2);
  __syncthreads();
  if (t < 7){
    float o = eb2[t];
    for (int k = 0; k < 128; ++k) o += h1[k] * ew2[k*7 + t];
    out[512 + b*7 + t] = o;
  }
  out[960 + b*256 + t]       = emb[t];
  out[960 + b*256 + 128 + t] = emb[128 + t];
}

extern "C" void kernel_launch(void* const* d_in, const int* in_sizes, int n_in,
                              void* d_out, int out_size, void* d_ws, size_t ws_size,
                              hipStream_t stream){
  const float* x_esm      = (const float*)d_in[0];
  const float* x_reschem  = (const float*)d_in[1];
  const float* x_role     = (const float*)d_in[2];
  const float* x_dist_raw = (const float*)d_in[3];
  const float* x_misc     = (const float*)d_in[4];
  const float* x_vec      = (const float*)d_in[5];
  const float* pos        = (const float*)d_in[6];
  const float* edge_dist  = (const float*)d_in[7];
  const float* ne_esm_w   = (const float*)d_in[8];
  const float* ne_esm_b   = (const float*)d_in[9];
  const float* ne_esm_ln_g= (const float*)d_in[10];
  const float* ne_esm_ln_b= (const float*)d_in[11];
  const float* ne_out_w   = (const float*)d_in[12];
  const float* ne_out_b   = (const float*)d_in[13];
  const float* ne_out_ln_g= (const float*)d_in[14];
  const float* ne_out_ln_b= (const float*)d_in[15];
  const float* ee_w       = (const float*)d_in[16];
  const float* ee_b       = (const float*)d_in[17];
  const float* ee_ln_g    = (const float*)d_in[18];
  const float* ee_ln_b    = (const float*)d_in[19];
  const float* init_vec_w = (const float*)d_in[20];
  const float* msg_s_w1   = (const float*)d_in[21];
  const float* msg_s_b1   = (const float*)d_in[22];
  const float* msg_s_w2   = (const float*)d_in[23];
  const float* msg_s_b2   = (const float*)d_in[24];
  const float* msg_v_w    = (const float*)d_in[25];
  const float* msg_g_w    = (const float*)d_in[26];
  const float* msg_g_b    = (const float*)d_in[27];
  const float* upd_s_w1   = (const float*)d_in[28];
  const float* upd_s_b1   = (const float*)d_in[29];
  const float* upd_s_w2   = (const float*)d_in[30];
  const float* upd_s_b2   = (const float*)d_in[31];
  const float* upd_v_w    = (const float*)d_in[32];
  const float* upd_g_w    = (const float*)d_in[33];
  const float* upd_g_b    = (const float*)d_in[34];
  const float* ln_g       = (const float*)d_in[35];
  const float* ln_b       = (const float*)d_in[36];
  const float* hm_w1      = (const float*)d_in[37];
  const float* hm_b1      = (const float*)d_in[38];
  const float* hm_w2      = (const float*)d_in[39];
  const float* hm_b2      = (const float*)d_in[40];
  const float* he_w1      = (const float*)d_in[41];
  const float* he_b1      = (const float*)d_in[42];
  const float* he_w2      = (const float*)d_in[43];
  const float* he_b2      = (const float*)d_in[44];
  const int*   edge_index = (const int*)d_in[45];
  const int*   batch      = (const int*)d_in[46];
  float* out = (float*)d_out;

  // workspace carve-up
  char* wsp = (char*)d_ws;
  auto carve = [&](size_t bytes)->char*{
    char* p = wsp; wsp += (bytes + 255) & ~(size_t)255; return p;
  };
  _Float16* Wesm  = (_Float16*)carve((size_t)ESMD*128*2);
  _Float16* Wout  = (_Float16*)carve((size_t)OUT_K*128*2);
  _Float16* Wmsg1 = (_Float16*)carve((size_t)NLAYER*MSG_K*128*2);
  _Float16* Wmsg2 = (_Float16*)carve((size_t)NLAYER*128*128*2);
  _Float16* Wupd1 = (_Float16*)carve((size_t)NLAYER*UPD_K*128*2);
  _Float16* Wupd2 = (_Float16*)carve((size_t)NLAYER*128*128*2);
  float*    tmp_esm = (float*)carve((size_t)NNODES*128*4);
  _Float16* xcat    = (_Float16*)carve((size_t)NNODES*OUT_K*2);
  float*    sbuf    = (float*)carve((size_t)NNODES*128*4);
  float*    vbuf    = (float*)carve((size_t)NNODES*48*4);
  _Float16* edge_s  = (_Float16*)carve((size_t)NEDGES*64*2);
  float*    edge_v  = (float*)carve((size_t)NEDGES*3*4);
  float*    edge_len= (float*)carve((size_t)NEDGES*4);
  float*    agg_s   = (float*)carve((size_t)NNODES*128*4);
  float*    agg_v   = (float*)carve((size_t)NNODES*48*4);
  float*    psum    = (float*)carve((size_t)NBATCH*128*4);
  float*    pcnt    = (float*)carve((size_t)NBATCH*4);
  unsigned* pmax    = (unsigned*)carve((size_t)NBATCH*128*4);

  dim3 blk(256);
  auto gfor = [](long n){ return dim3((unsigned)((n + 255) / 256)); };

  // ---- weight packing (f32 -> f16 WMMA B-fragment layout) ----
  pack_b<<<gfor((long)ESMD*128), blk, 0, stream>>>(ne_esm_w, Wesm, ESMD, 128, ESMD);
  pack_b<<<gfor((long)OUT_K*128), blk, 0, stream>>>(ne_out_w, Wout, 203, 128, OUT_K);
  for (int l = 0; l < NLAYER; ++l){
    pack_b<<<gfor((long)MSG_K*128), blk, 0, stream>>>(msg_s_w1 + (long)l*354*128, Wmsg1 + (long)l*MSG_K*128, 354, 128, MSG_K);
    pack_b<<<gfor((long)128*128), blk, 0, stream>>>(msg_s_w2 + (long)l*128*128, Wmsg2 + (long)l*128*128, 128, 128, 128);
    pack_b<<<gfor((long)UPD_K*128), blk, 0, stream>>>(upd_s_w1 + (long)l*UPD_K*128, Wupd1 + (long)l*UPD_K*128, UPD_K, 128, UPD_K);
    pack_b<<<gfor((long)128*128), blk, 0, stream>>>(upd_s_w2 + (long)l*128*128, Wupd2 + (long)l*128*128, 128, 128, 128);
  }

  // ---- node & edge encoders ----
  gemm_esm<<<dim3(NNODES/16), blk, 0, stream>>>(x_esm, Wesm, tmp_esm);
  node_feat<<<dim3(NNODES/8), blk, 0, stream>>>(tmp_esm, ne_esm_b, ne_esm_ln_g, ne_esm_ln_b,
                                                x_reschem, x_role, x_misc, x_dist_raw, xcat);
  gemm_node_out<<<dim3(NNODES/16), blk, 0, stream>>>(xcat, Wout, ne_out_b, ne_out_ln_g, ne_out_ln_b, sbuf);
  init_v<<<gfor((long)NNODES*48), blk, 0, stream>>>(x_vec, init_vec_w, vbuf);
  edge_feat<<<dim3(NEDGES/8), blk, 0, stream>>>(edge_dist, edge_index, pos, ee_w, ee_b,
                                                ee_ln_g, ee_ln_b, edge_s, edge_v, edge_len);

  // ---- message passing layers ----
  for (int l = 0; l < NLAYER; ++l){
    zero_f32<<<gfor((long)NNODES*128), blk, 0, stream>>>(agg_s, (long)NNODES*128);
    zero_f32<<<gfor((long)NNODES*48), blk, 0, stream>>>(agg_v, (long)NNODES*48);
    msg_kernel<<<dim3(NEDGES/16), blk, 0, stream>>>(
        sbuf, vbuf, edge_s, edge_v, edge_len, edge_index,
        Wmsg1 + (long)l*MSG_K*128, msg_s_b1 + (long)l*128,
        Wmsg2 + (long)l*128*128,   msg_s_b2 + (long)l*128,
        msg_v_w + (long)l*33*16, msg_g_w + (long)l*128*16, msg_g_b + (long)l*16,
        agg_s, agg_v);
    upd_kernel<<<dim3(NNODES/16), blk, 0, stream>>>(
        sbuf, vbuf, agg_s, agg_v,
        Wupd1 + (long)l*UPD_K*128, upd_s_b1 + (long)l*128,
        Wupd2 + (long)l*128*128,   upd_s_b2 + (long)l*128,
        upd_v_w + (long)l*32*16, upd_g_w + (long)l*128*16, upd_g_b + (long)l*16,
        ln_g + (long)l*128, ln_b + (long)l*128);
  }

  // ---- pooling + heads ----
  zero_f32<<<gfor((long)NBATCH*128), blk, 0, stream>>>(psum, (long)NBATCH*128);
  zero_f32<<<gfor((long)NBATCH), blk, 0, stream>>>(pcnt, (long)NBATCH);
  zero_f32<<<gfor((long)NBATCH*128), blk, 0, stream>>>((float*)pmax, (long)NBATCH*128);
  pool_kernel<<<dim3(NNODES/32), blk, 0, stream>>>(sbuf, batch, psum, pcnt, pmax);
  head_kernel<<<dim3(NBATCH), dim3(128), 0, stream>>>(psum, pcnt, pmax,
      hm_w1, hm_b1, hm_w2, hm_b2, he_w1, he_b1, he_w2, he_b2, out);
}
